// GAT_Base_53575422050306
// MI455X (gfx1250) — matrix operand order repaired and checked by reading, not compiled, not run on hardware
//
#include <hip/hip_runtime.h>
#include <hip/hip_bf16.h>

typedef __attribute__((ext_vector_type(16))) _Float16 v16h;
typedef __attribute__((ext_vector_type(8)))  _Float16 v8h;
typedef __attribute__((ext_vector_type(8)))  float    v8f;

#define TILE_M 128
#define TILE_N 64
#define LDA  40   // f16 elems; 80 B row stride -> 16B aligned, conflict-free
#define LDBT 40   // f16 elems; K-contiguous per column

union H16 { v16h v; v8h p[2]; _Float16 h[16]; };
union H8  { v8h  v; _Float16 h[8]; };
union F8  { v8f  v; float    f[8]; };

// ---------------------------------------------------------------------------
// WMMA GEMM: C[M,N] = A[M,K] @ B[K,N]; f32 memory, f16 WMMA, f32 accumulate.
// Block = 256 threads = 8 waves; block tile 128x64. Each wave owns a 16-row
// stripe and all four 16-wide column tiles (1 A fragment -> 4 WMMAs / K-step).
// Bs staged K-contiguous per column so every fragment is two ds_load_b128.
// ---------------------------------------------------------------------------
__global__ __launch_bounds__(256)
void gemm_wmma(const float* __restrict__ A, const float* __restrict__ B,
               float* __restrict__ C, int M, int K, int N) {
    __shared__ __attribute__((aligned(16))) _Float16 As[TILE_M * LDA];
    __shared__ __attribute__((aligned(16))) _Float16 Bs[TILE_N * LDBT];

    const int t     = threadIdx.x;
    const int warp  = t >> 5;
    const int lane  = t & 31;
    const int halfw = lane >> 4;      // 0: lanes 0-15, 1: lanes 16-31
    const int mr    = lane & 15;
    const int blockRow = blockIdx.x * TILE_M;
    const int colbase  = blockIdx.y * TILE_N;

    F8 acc[4];
#pragma unroll
    for (int nn = 0; nn < 4; ++nn)
#pragma unroll
        for (int i = 0; i < 8; ++i) acc[nn].f[i] = 0.f;

    const int tr   = t >> 3;          // 0..31   (A staging row)
    const int tc4  = (t & 7) << 2;    // 0..28   (A staging col, x4 f32)
    const int bcol = t & 63;          // 0..63   (B staging column)
    const int bk   = (t >> 6) << 3;   // 0,8,16,24 (B staging K chunk)

    for (int k0 = 0; k0 < K; k0 += 32) {
        // ---- stage A block: 128x32 f32 -> f16 LDS (row-major) ----
#pragma unroll
        for (int rr = 0; rr < 4; ++rr) {
            int row  = tr + (rr << 5);
            int grow = blockRow + row;
            float4 va = make_float4(0.f, 0.f, 0.f, 0.f);
            if (grow < M) {
                const float* ap = A + (size_t)grow * K + k0 + tc4;
                va = *(const float4*)ap;
                __builtin_prefetch(ap + 32, 0, 3);   // next K tile
            }
            _Float16* as = &As[row * LDA + tc4];
            as[0] = (_Float16)va.x; as[1] = (_Float16)va.y;
            as[2] = (_Float16)va.z; as[3] = (_Float16)va.w;
        }
        // ---- stage B block: 32x64 f32 -> f16 LDS, K-contiguous per column ----
        {
            H8 bpack;
#pragma unroll
            for (int j = 0; j < 8; ++j)   // coalesced: lanes sweep columns
                bpack.h[j] = (_Float16)B[(size_t)(k0 + bk + j) * N + colbase + bcol];
            *(v8h*)&Bs[bcol * LDBT + bk] = bpack.v;   // one ds_store_b128
        }
        __syncthreads();

        // ---- fragments: all vector LDS loads ----
        H16 af;
        const _Float16* arow = &As[(warp * 16 + mr) * LDA];
        af.p[0] = *(const v8h*)&arow[8 * halfw];        // K = 8h .. 8h+7
        af.p[1] = *(const v8h*)&arow[16 + 8 * halfw];   // K = 16+8h .. 16+8h+7

        const int kb = halfw * 8;                        // B: K = kb..kb+7, 16+kb..16+kb+7
#pragma unroll
        for (int nn = 0; nn < 4; ++nn) {
            H16 bf;
            const _Float16* bp = &Bs[(nn * 16 + mr) * LDBT + kb];
            bf.p[0] = *(const v8h*)bp;
            bf.p[1] = *(const v8h*)(bp + 16);
            acc[nn].v = __builtin_amdgcn_wmma_f32_16x16x32_f16(
                false, af.v, false, bf.v, (short)0, acc[nn].v, false, false);
        }
        __syncthreads();
    }

    // ---- store C (ISA 16x16 f32 C/D lane layout) ----
#pragma unroll
    for (int nn = 0; nn < 4; ++nn)
#pragma unroll
        for (int r = 0; r < 8; ++r) {
            int grow = blockRow + warp * 16 + r + 8 * halfw;
            if (grow < M)
                C[(size_t)grow * N + colbase + nn * 16 + mr] = acc[nn].f[r];
        }
}

// ---------------------------------------------------------------------------
// Wave per (node, head): coalesced reduction of h[n,h,:] . a_src / a_dst
// ---------------------------------------------------------------------------
__global__ void alpha_kernel(const float* __restrict__ h,
                             const float* __restrict__ a_src,
                             const float* __restrict__ a_dst,
                             float* __restrict__ out_s, float* __restrict__ out_d,
                             int Nn, int H, int C) {
    int gid  = blockIdx.x * blockDim.x + threadIdx.x;
    int wid  = gid >> 5, lane = gid & 31;
    if (wid >= Nn * H) return;
    int n = wid / H, hd = wid % H;
    const float* hp = h + (size_t)n * H * C + (size_t)hd * C;
    const float* a1 = a_src + hd * C;
    const float* a2 = a_dst + hd * C;
    float v1 = hp[lane], v2 = hp[lane + 32];
    float s1 = v1 * a1[lane] + v2 * a1[lane + 32];
    float s2 = v1 * a2[lane] + v2 * a2[lane + 32];
#pragma unroll
    for (int off = 16; off > 0; off >>= 1) {
        s1 += __shfl_xor(s1, off, 32);
        s2 += __shfl_xor(s2, off, 32);
    }
    if (lane == 0) { out_s[wid] = s1; out_d[wid] = s2; }
}

__global__ void fill_kernel(float* __restrict__ p, float v, size_t n) {
    size_t i = (size_t)blockIdx.x * blockDim.x + threadIdx.x;
    if (i < n) p[i] = v;
}

__device__ inline void atomicMaxF(float* addr, float val) {
    int* ai = (int*)addr;
    int old = __float_as_int(*addr);
    while (val > __int_as_float(old)) {
        int assumed = old;
        old = atomicCAS(ai, assumed, __float_as_int(val));
        if (old == assumed) break;
    }
}

// pass1: e = leaky_relu(as[src]+ad[dst]); store e; segment max into mmax[dst]
__global__ void edge_pass1(const long long* __restrict__ ei, long long E, int Nn,
                           int hshift,
                           const float* __restrict__ as_, const float* __restrict__ ad_,
                           float* __restrict__ ebuf, float* __restrict__ mmax) {
    const int H = 1 << hshift, hm = H - 1;
    long long Et = E + Nn;
    long long i = (long long)blockIdx.x * blockDim.x + threadIdx.x;
    if (i >= (Et << hshift)) return;
    long long e = i >> hshift; int hd = (int)(i & hm);
    int s = (e < E) ? (int)ei[e]     : (int)(e - E);
    int d = (e < E) ? (int)ei[E + e] : (int)(e - E);
    float v = as_[(size_t)s * H + hd] + ad_[(size_t)d * H + hd];
    v = (v > 0.f) ? v : 0.2f * v;
    ebuf[i] = v;
    atomicMaxF(&mmax[(size_t)d * H + hd], v);
}

// pass2: ex = exp(e - m[dst]); store; segment sum into den[dst]
__global__ void edge_pass2(const long long* __restrict__ ei, long long E, int Nn,
                           int hshift,
                           float* __restrict__ ebuf, const float* __restrict__ mmax,
                           float* __restrict__ den) {
    const int H = 1 << hshift, hm = H - 1;
    long long Et = E + Nn;
    long long i = (long long)blockIdx.x * blockDim.x + threadIdx.x;
    if (i >= (Et << hshift)) return;
    long long e = i >> hshift; int hd = (int)(i & hm);
    int d = (e < E) ? (int)ei[E + e] : (int)(e - E);
    float ex = __expf(ebuf[i] - mmax[(size_t)d * H + hd]);
    ebuf[i] = ex;
    atomicAdd(&den[(size_t)d * H + hd], ex);
}

// pass3: one wave per (edge, head); 64 channels -> 2 per lane; scatter-add.
__global__ void edge_pass3(const long long* __restrict__ ei, long long E, int Nn,
                           int hshift,
                           const float* __restrict__ ebuf, const float* __restrict__ den,
                           const float* __restrict__ hsrc, float* __restrict__ out) {
    const int C = 64;
    const int H = 1 << hshift, hm = H - 1;
    long long gid = (long long)blockIdx.x * blockDim.x + threadIdx.x;
    long long wid = gid >> 5; int lane = (int)(gid & 31);
    long long Et = E + Nn;
    if (wid >= (Et << hshift)) return;
    long long e = wid >> hshift; int hd = (int)(wid & hm);
    int s = (e < E) ? (int)ei[e]     : (int)(e - E);
    int d = (e < E) ? (int)ei[E + e] : (int)(e - E);
    float alpha = ebuf[wid] / (den[(size_t)d * H + hd] + 1e-16f);
    const float* hp = hsrc + (size_t)s * H * C + (size_t)hd * C;
    float*       op = out  + (size_t)d * H * C + (size_t)hd * C;
    int c0 = lane * 2;
    atomicAdd(&op[c0],     hp[c0]     * alpha);
    atomicAdd(&op[c0 + 1], hp[c0 + 1] * alpha);
}

__global__ void bias_relu(float* __restrict__ p, const float* __restrict__ b,
                          size_t n, int dmask) {
    size_t i = (size_t)blockIdx.x * blockDim.x + threadIdx.x;
    if (i < n) {
        float v = p[i] + b[i & dmask];
        p[i] = v > 0.f ? v : 0.f;
    }
}

__global__ void pool_kernel(const float* __restrict__ h, const long long* __restrict__ batch,
                            float* __restrict__ sums, float* __restrict__ cnts, int Nn) {
    int i = blockIdx.x * blockDim.x + threadIdx.x;
    if (i >= Nn * 64) return;
    int n = i >> 6, c = i & 63;
    int g = (int)batch[n];
    atomicAdd(&sums[g * 64 + c], h[(size_t)n * 64 + c]);
    if (c == 0) atomicAdd(&cnts[g], 1.0f);
}

__global__ void classifier_kernel(const float* __restrict__ sums, const float* __restrict__ cnts,
                                  const float* __restrict__ Wl, const float* __restrict__ bl,
                                  float* __restrict__ out) {
    int i = blockIdx.x * blockDim.x + threadIdx.x;
    if (i >= 64 * 10) return;
    int g = i / 10, c = i % 10;
    float cnt = cnts[g]; cnt = cnt > 1.f ? cnt : 1.f;
    float acc = 0.f;
    for (int k = 0; k < 64; ++k) acc += (sums[g * 64 + k] / cnt) * Wl[k * 10 + c];
    out[i] = acc + bl[c];
}

// ---------------------------------------------------------------------------
static void run_gat_layer(const float* h_in, const float* W,
                          const float* a_src, const float* a_dst, const float* bias,
                          const long long* ei, long long E, int Nn, int K, int hshift,
                          float* proj, float* out,
                          float* as_, float* ad_, float* mmax, float* den, float* ebuf,
                          hipStream_t stream) {
    const int C = 64, H = 1 << hshift, D = H * C;
    dim3 gg((Nn + TILE_M - 1) / TILE_M, D / TILE_N);
    gemm_wmma<<<gg, 256, 0, stream>>>(h_in, W, proj, Nn, K, D);

    int nh = Nn * H;
    alpha_kernel<<<((size_t)nh * 32 + 255) / 256, 256, 0, stream>>>(
        proj, a_src, a_dst, as_, ad_, Nn, H, C);
    fill_kernel<<<(nh + 255) / 256, 256, 0, stream>>>(mmax, -3.0e38f, (size_t)nh);
    fill_kernel<<<(nh + 255) / 256, 256, 0, stream>>>(den, 0.f, (size_t)nh);

    long long Et = E + Nn;
    long long w12 = Et << hshift;
    edge_pass1<<<(unsigned)((w12 + 255) / 256), 256, 0, stream>>>(
        ei, E, Nn, hshift, as_, ad_, ebuf, mmax);
    edge_pass2<<<(unsigned)((w12 + 255) / 256), 256, 0, stream>>>(
        ei, E, Nn, hshift, ebuf, mmax, den);

    size_t nd = (size_t)Nn * D;
    fill_kernel<<<(unsigned)((nd + 255) / 256), 256, 0, stream>>>(out, 0.f, nd);
    long long w3 = w12 * 32;
    edge_pass3<<<(unsigned)((w3 + 255) / 256), 256, 0, stream>>>(
        ei, E, Nn, hshift, ebuf, den, proj, out);
    bias_relu<<<(unsigned)((nd + 255) / 256), 256, 0, stream>>>(out, bias, nd, D - 1);
}

extern "C" void kernel_launch(void* const* d_in, const int* in_sizes, int n_in,
                              void* d_out, int out_size, void* d_ws, size_t ws_size,
                              hipStream_t stream) {
    const float*     x     = (const float*)d_in[0];
    const long long* ei    = (const long long*)d_in[1];
    const long long* batch = (const long long*)d_in[2];
    const float* W1  = (const float*)d_in[3];
    const float* as1 = (const float*)d_in[4];
    const float* ad1 = (const float*)d_in[5];
    const float* b1  = (const float*)d_in[6];
    const float* W2  = (const float*)d_in[7];
    const float* as2 = (const float*)d_in[8];
    const float* ad2 = (const float*)d_in[9];
    const float* b2  = (const float*)d_in[10];
    const float* W3  = (const float*)d_in[11];
    const float* as3 = (const float*)d_in[12];
    const float* ad3 = (const float*)d_in[13];
    const float* b3  = (const float*)d_in[14];
    const float* Wl  = (const float*)d_in[15];
    const float* bl  = (const float*)d_in[16];
    float* out = (float*)d_out;

    const int Nn = in_sizes[0] / 128;       // 50000
    const long long E = in_sizes[1] / 2;    // 800000
    const int G = 64;

    // workspace layout (floats)
    float* ws   = (float*)d_ws;
    float* buf0 = ws;                                  // proj   [N,256]
    float* buf1 = buf0 + (size_t)Nn * 256;             // in/out [N,256]
    float* as_  = buf1 + (size_t)Nn * 256;             // [N,4]
    float* ad_  = as_  + (size_t)Nn * 4;               // [N,4]
    float* mmax = ad_  + (size_t)Nn * 4;               // [N,4]
    float* den  = mmax + (size_t)Nn * 4;               // [N,4]
    float* ebuf = den  + (size_t)Nn * 4;               // [(E+N),4]
    float* sums = ebuf + (size_t)(E + Nn) * 4;         // [64,64]
    float* cnts = sums + G * 64;                       // [64]

    // layer 1: x[N,128] -> buf1[N,256]   (H=4 -> hshift=2)
    run_gat_layer(x,    W1, as1, ad1, b1, ei, E, Nn, 128, 2,
                  buf0, buf1, as_, ad_, mmax, den, ebuf, stream);
    // layer 2: buf1[N,256] -> buf1[N,256]
    run_gat_layer(buf1, W2, as2, ad2, b2, ei, E, Nn, 256, 2,
                  buf0, buf1, as_, ad_, mmax, den, ebuf, stream);
    // layer 3: buf1[N,256] -> buf1[N,64]  (heads=1, concat=False == mean of 1)
    run_gat_layer(buf1, W3, as3, ad3, b3, ei, E, Nn, 256, 0,
                  buf0, buf1, as_, ad_, mmax, den, ebuf, stream);

    // global mean pool + classifier
    fill_kernel<<<(G * 64 + 255) / 256, 256, 0, stream>>>(sums, 0.f, (size_t)G * 64);
    fill_kernel<<<1, 64, 0, stream>>>(cnts, 0.f, (size_t)G);
    pool_kernel<<<(Nn * 64 + 255) / 256, 256, 0, stream>>>(buf1, batch, sums, cnts, Nn);
    classifier_kernel<<<(640 + 255) / 256, 256, 0, stream>>>(sums, cnts, Wl, bl, out);
}
